// GroupedExperts_23682449670360
// MI455X (gfx1250) — matrix operand order repaired and checked by reading, not compiled, not run on hardware
//
#include <hip/hip_runtime.h>

// Grouped-experts SwiGLU FFN for MI455X (gfx1250, wave32, WMMA).
//   per expert e: h = silu(X @ W1[e]) * (X @ W3[e]); out = h @ W2[e]
// fp32 in/out; bf16 WMMA (V_WMMA_F32_16X16X32_BF16) with f32 accumulation.
// B fragments are shared across all 4 M-tiles (nt-outer / K-middle / mt-inner).

typedef __attribute__((ext_vector_type(16))) __bf16 v16bf;
typedef __attribute__((ext_vector_type(8)))  float  v8f;

#define NUM_EXPERTS 64
#define DIM         2048
#define HID         1024
#define TPE         256     // tokens per expert
#define TILE_M      64      // token rows per block
#define MT          (TILE_M / 16)          // 4 M-tiles
#define TILES_PER_E (TPE / TILE_M)         // 4
#define WAVES       8

__device__ __forceinline__ __bf16 f2bf(float f) { return (__bf16)f; }

__global__ __launch_bounds__(256)
void moe_swiglu_wmma(const float* __restrict__ x,
                     const float* __restrict__ w1,
                     const float* __restrict__ w2,
                     const float* __restrict__ w3,
                     float* __restrict__ out)
{
    // Post-activation hidden tile, bf16: 64 x 1024 x 2B = 128 KB LDS.
    __shared__ __bf16 Hs[TILE_M * HID];

    const int bx     = blockIdx.x;
    const int expert = bx / TILES_PER_E;
    const int tile   = bx % TILES_PER_E;
    const int tok0   = expert * TPE + tile * TILE_M;

    const int tid  = threadIdx.x;
    const int wave = tid >> 5;             // 0..7
    const int lane = tid & 31;
    const int lmod = lane & 15;
    const bool hi  = lane >= 16;

    const float* __restrict__ Xb = x  + (size_t)tok0   * DIM;
    const float* __restrict__ W1 = w1 + (size_t)expert * DIM * HID;
    const float* __restrict__ W3 = w3 + (size_t)expert * DIM * HID;
    const float* __restrict__ W2 = w2 + (size_t)expert * HID * DIM;

    // ---------------- Phase A: H = silu(X@W1) * (X@W3) ----------------
    // Wave w owns hidden columns [w*128, (w+1)*128): MT x 8 tiles of 16x16.
    for (int nt = 0; nt < 8; ++nt) {
        const int ncol = wave * 128 + nt * 16 + lmod;   // B-frag column
        v8f acc1[MT] = {};
        v8f acc3[MT] = {};
        for (int k0 = 0; k0 < DIM; k0 += 32) {
            // A fragments (one per M-tile): 16-bit A 16x32 layout -> lane
            // holds K = {ka..ka+7} and {ka+16..ka+23}, ka = k0 (+8 hi lanes)
            const int ka = k0 + (hi ? 8 : 0);
            v16bf a[MT];
            #pragma unroll
            for (int mt = 0; mt < MT; ++mt) {
                const float* __restrict__ xr =
                    Xb + (size_t)(mt * 16 + lmod) * DIM;
                #pragma unroll
                for (int j = 0; j < 8; ++j) a[mt][j]     = f2bf(xr[ka + j]);
                #pragma unroll
                for (int j = 0; j < 8; ++j) a[mt][8 + j] = f2bf(xr[ka + 16 + j]);
            }
            // B fragments: lane holds K = {kb..kb+7} and {kb+16..kb+23} for one column,
            // kb = k0 (+8 hi lanes). W1/W3 are [DIM][HID] row-major.
            // Loaded ONCE and reused by all MT M-tiles.
            const int kb = k0 + (hi ? 8 : 0);
            v16bf b1, b3;
            #pragma unroll
            for (int j = 0; j < 16; ++j) {
                const size_t off = (size_t)(kb + j + (j >= 8 ? 8 : 0)) * HID + ncol;
                b1[j] = f2bf(W1[off]);
                b3[j] = f2bf(W3[off]);
            }
            #pragma unroll
            for (int mt = 0; mt < MT; ++mt) {
                acc1[mt] = __builtin_amdgcn_wmma_f32_16x16x32_bf16(
                    false, a[mt], false, b1, (short)0, acc1[mt], false, false);
                acc3[mt] = __builtin_amdgcn_wmma_f32_16x16x32_bf16(
                    false, a[mt], false, b3, (short)0, acc3[mt], false, false);
            }
        }
        // SwiGLU epilogue: silu(g)*u with fast v_rcp_f32 (no IEEE div chain).
        // C/D layout: VGPR r -> row mt*16 + r (+8 hi lanes), col = lmod.
        #pragma unroll
        for (int mt = 0; mt < MT; ++mt) {
            #pragma unroll
            for (int r = 0; r < 8; ++r) {
                const float g = acc1[mt][r];
                const float u = acc3[mt][r];
                const float s = g * __builtin_amdgcn_rcpf(1.0f + __expf(-g));
                const int row = mt * 16 + r + (hi ? 8 : 0);
                const int col = wave * 128 + nt * 16 + lmod;
                Hs[row * HID + col] = f2bf(s * u);
            }
        }
    }

    __syncthreads();

    // ---------------- Phase B: out = H @ W2 ----------------
    // Wave w owns output columns [w*256, (w+1)*256): MT x 16 tiles of 16x16.
    for (int nt = 0; nt < 16; ++nt) {
        const int ncol = wave * 256 + nt * 16 + lmod;
        v8f acc[MT] = {};
        for (int k0 = 0; k0 < HID; k0 += 32) {
            const int ka = k0 + (hi ? 8 : 0);
            v16bf a[MT];
            #pragma unroll
            for (int mt = 0; mt < MT; ++mt) {
                const __bf16* __restrict__ hr = &Hs[(mt * 16 + lmod) * HID];
                // Two 16-byte LDS loads: K {ka..ka+7}, {ka+16..ka+23}.
                *(uint4*)&a[mt]       = *(const uint4*)(hr + ka);
                *((uint4*)&a[mt] + 1) = *(const uint4*)(hr + ka + 16);
            }
            const int kb = k0 + (hi ? 8 : 0);
            v16bf b;       // W2 is [HID][DIM] row-major; shared by all M-tiles
            #pragma unroll
            for (int j = 0; j < 16; ++j)
                b[j] = f2bf(W2[(size_t)(kb + j + (j >= 8 ? 8 : 0)) * DIM + ncol]);

            #pragma unroll
            for (int mt = 0; mt < MT; ++mt)
                acc[mt] = __builtin_amdgcn_wmma_f32_16x16x32_bf16(
                    false, a[mt], false, b, (short)0, acc[mt], false, false);
        }
        #pragma unroll
        for (int mt = 0; mt < MT; ++mt) {
            #pragma unroll
            for (int r = 0; r < 8; ++r) {
                const int row = mt * 16 + r + (hi ? 8 : 0);
                out[(size_t)(tok0 + row) * DIM + ncol] = acc[mt][r];
            }
        }
    }
}

extern "C" void kernel_launch(void* const* d_in, const int* in_sizes, int n_in,
                              void* d_out, int out_size, void* d_ws, size_t ws_size,
                              hipStream_t stream) {
    // setup_inputs order: x, w1, w2, w3, num_local_tokens_per_expert
    const float* x  = (const float*)d_in[0];
    const float* w1 = (const float*)d_in[1];
    const float* w2 = (const float*)d_in[2];
    const float* w3 = (const float*)d_in[3];
    float* out = (float*)d_out;

    dim3 grid(NUM_EXPERTS * TILES_PER_E);   // 256 blocks
    dim3 block(256);                        // 8 wave32s
    moe_swiglu_wmma<<<grid, block, 0, stream>>>(x, w1, w2, w3, out);
}
